// GPT_43954695308125
// MI455X (gfx1250) — matrix operand order repaired and checked by reading, not compiled, not run on hardware
//
#include <hip/hip_runtime.h>
#include <hip/hip_bf16.h>

// ---------------------------------------------------------------------------
// MI455X (gfx1250) GPT forward: WMMA f16 GEMMs with async-to-LDS double
// buffering + flash-style WMMA attention. wave32; WMMA 16x16x32 f16 -> f32.
// ---------------------------------------------------------------------------

typedef __attribute__((ext_vector_type(16))) _Float16 v16h;
typedef __attribute__((ext_vector_type(8)))  _Float16 v8h;
typedef __attribute__((ext_vector_type(8)))  float    v8f;

#define WMMA16(a,b,c) __builtin_amdgcn_wmma_f32_16x16x32_f16(false,(a),false,(b),(short)0,(c),false,false)

// Build a v16h A/B fragment from two contiguous 8-half (16B) chunks.
static __device__ __forceinline__ v16h mk16(const _Float16* p0, const _Float16* p1) {
  v8h lo = *(const v8h*)p0;
  v8h hi = *(const v8h*)p1;
  v16h r;
#pragma unroll
  for (int i = 0; i < 8; ++i) { r[i] = lo[i]; r[i + 8] = hi[i]; }
  return r;
}

static __device__ __forceinline__ void async_b128(unsigned lds, const _Float16* g) {
  unsigned long long ga = (unsigned long long)(uintptr_t)g;
  asm volatile("global_load_async_to_lds_b128 %0, %1, off"
               :: "v"(lds), "v"(ga) : "memory");
}

// ---------------------------------------------------------------------------
// Embedding gather: x[s,d] = emb[idx[s], d]   (f32)
// ---------------------------------------------------------------------------
__global__ void embed_kernel(const int* __restrict__ idx, const float* __restrict__ emb,
                             float* __restrict__ x, int S, int D) {
  int i = blockIdx.x * blockDim.x + threadIdx.x;
  if (i < S * D) {
    int s = i / D, d = i % D;
    x[i] = emb[(size_t)idx[s] * D + d];
  }
}

// ---------------------------------------------------------------------------
// RMSNorm row -> f16: h[row,:] = x[row,:] * rsqrt(mean(x^2)+eps)
// ---------------------------------------------------------------------------
__global__ __launch_bounds__(256) void rmsnorm_f16(const float* __restrict__ x,
                                                   _Float16* __restrict__ h, int n) {
  __shared__ float red[256];
  int row = blockIdx.x;
  const float* p = x + (size_t)row * n;
  float ss = 0.f;
  for (int i = threadIdx.x; i < n; i += 256) { float v = p[i]; ss += v * v; }
  red[threadIdx.x] = ss;
  __syncthreads();
  for (int d = 128; d > 0; d >>= 1) {
    if (threadIdx.x < d) red[threadIdx.x] += red[threadIdx.x + d];
    __syncthreads();
  }
  float r = rsqrtf(red[0] / (float)n + 1.1920929e-7f);
  for (int i = threadIdx.x; i < n; i += 256) h[(size_t)row * n + i] = (_Float16)(p[i] * r);
}

// ---------------------------------------------------------------------------
// Per-row int8 fake-quant -> f16 weights.
// quantile(|w|, 0.9999984) interpolates between top-2 order statistics.
// ---------------------------------------------------------------------------
__global__ __launch_bounds__(256) void quant_rows(const float* __restrict__ W,
                                                  _Float16* __restrict__ out, int ncols) {
  __shared__ float r1[256], r2[256];
  __shared__ float shc[3];
  int row = blockIdx.x, tid = threadIdx.x;
  const float* p = W + (size_t)row * ncols;
  float m1 = 0.f, m2 = 0.f;
  for (int i = tid; i < ncols; i += 256) {
    float a = fabsf(p[i]);
    if (a > m1) { m2 = m1; m1 = a; } else if (a > m2) { m2 = a; }
  }
  r1[tid] = m1; r2[tid] = m2;
  __syncthreads();
  for (int d = 128; d > 0; d >>= 1) {
    if (tid < d) {
      float a1 = r1[tid], a2 = r2[tid], b1 = r1[tid + d], b2 = r2[tid + d];
      r1[tid] = fmaxf(a1, b1);
      r2[tid] = fmaxf(fminf(a1, b1), fmaxf(a2, b2));
    }
    __syncthreads();
  }
  if (tid == 0) {
    double frac = 0.9999984 * (double)(ncols - 1) - (double)(ncols - 2);
    float clip = r2[0] + (float)frac * (r1[0] - r2[0]);
    float scale = fmaxf(clip / 127.f, 1.f / 127.f);
    float s16 = (float)(_Float16)scale;   // f16 round-trip of the scale
    shc[0] = clip; shc[1] = scale; shc[2] = s16;
  }
  __syncthreads();
  float clip = shc[0], scale = shc[1], s16 = shc[2];
  for (int i = tid; i < ncols; i += 256) {
    float w = p[i];
    float c = fminf(fmaxf(w, -clip), clip);
    float q = rintf(c / scale);              // round-half-even, matches jnp.round
    q = fminf(fmaxf(q, -127.f), 127.f);
    out[(size_t)row * ncols + i] = (_Float16)(q * s16);
  }
}

// ---------------------------------------------------------------------------
// WMMA GEMM: O[M,N] = A[M,K](f16) @ W[N,K]^T(f16), f32 accumulate.
// Block = 8 waves; block tile 128(M) x 64(N); K-step 64 (8 WMMAs per stage).
// B tiles staged into LDS with global_load_async_to_lds_b128 (ASYNCcnt),
// double-buffered so the next tile's DMA overlaps the current tile's WMMAs.
// A fragments are register double-buffered.
// ep: 0 = store f32; 1 = f32 residual add in place; 2 = relu^2 -> f16
// ---------------------------------------------------------------------------
__global__ __launch_bounds__(256) void gemm_wmma(const _Float16* __restrict__ A,
                                                 const _Float16* __restrict__ W,
                                                 float* __restrict__ outf,
                                                 _Float16* __restrict__ outh,
                                                 int M, int N, int K, int ep) {
  // double-buffered B tile: [buf][64 cols][64 K-halves]  (16 KB total)
  __shared__ __align__(16) _Float16 bbuf[2 * 64 * 64];

  int tid = threadIdx.x;
  int lane = tid & 31, wv = tid >> 5;
  int ln = lane & 15, lh = lane >> 4;
  int mbase = blockIdx.y * 128 + wv * 16;
  int nbase = blockIdx.x * 64;

  // cooperative async stage of one 64x64 B tile (each thread moves 32B)
  int scol = tid >> 2;                  // 0..63
  int schunk = (tid & 3) * 16;          // half offset within K-step (16 halves)
  int gcol = nbase + scol; if (gcol >= N) gcol = N - 1;
  const _Float16* gbase = W + (size_t)gcol * K + schunk;
  unsigned ldsb = (unsigned)(uintptr_t)&bbuf[(size_t)scol * 64 + schunk];

  v8f zero = {};
  v8f acc[4];
#pragma unroll
  for (int t = 0; t < 4; ++t) acc[t] = zero;

  const _Float16* arow = A + (size_t)(mbase + ln) * K;

  // stage first tile (kk = 0): two async b128 ops per thread
  async_b128(ldsb, gbase);
  async_b128(ldsb + 16, gbase + 8);

  // preload first A fragments (k 0..31 and 32..63)
  v16h a0 = mk16(arow + 8 * lh, arow + 16 + 8 * lh);
  v16h a1 = mk16(arow + 32 + 8 * lh, arow + 48 + 8 * lh);

  int nsteps = K >> 6;
  for (int s = 0; s < nsteps; ++s) {
    int kk = s << 6;
    int cur = s & 1;

    __syncthreads();   // readers of the buffer we are about to overwrite are done
    if (s + 1 < nsteps) {
      const _Float16* gn = gbase + kk + 64;
      unsigned ldsn = ldsb ^ (unsigned)(64 * 64 * sizeof(_Float16)); // other buffer
      async_b128(ldsn, gn);
      async_b128(ldsn + 16, gn + 8);
      asm volatile("s_wait_asynccnt 0x2" ::: "memory");  // current tile's DMA done
    } else {
      asm volatile("s_wait_asynccnt 0x0" ::: "memory");
    }
    __syncthreads();   // current tile staged by all waves

    // register-pipeline next A fragments (their wait overlaps WMMAs below)
    v16h a0n = a0, a1n = a1;
    if (s + 1 < nsteps) {
      a0n = mk16(arow + kk + 64 + 8 * lh, arow + kk + 80 + 8 * lh);
      a1n = mk16(arow + kk + 96 + 8 * lh, arow + kk + 112 + 8 * lh);
    }

    const _Float16* bt = &bbuf[(size_t)cur * 64 * 64];
#pragma unroll
    for (int t = 0; t < 4; ++t) {
      const _Float16* bp = bt + (size_t)(t * 16 + ln) * 64 + 8 * lh;
      v16h b0 = mk16(bp, bp + 16);            // k 0..31 of this step
      v16h b1 = mk16(bp + 32, bp + 48);      // k 32..63 of this step
      acc[t] = WMMA16(a0, b0, acc[t]);
      acc[t] = WMMA16(a1, b1, acc[t]);
    }
    a0 = a0n; a1 = a1n;
  }

  int rowoff = lh * 8;
#pragma unroll
  for (int t = 0; t < 4; ++t) {
#pragma unroll
    for (int i = 0; i < 8; ++i) {
      int row = mbase + i + rowoff;
      int col = nbase + t * 16 + ln;
      if (col < N) {
        float v = acc[t][i];
        if (ep == 0)      outf[(size_t)row * N + col] = v;
        else if (ep == 1) outf[(size_t)row * N + col] += v;
        else { v = fmaxf(v, 0.f); outh[(size_t)row * N + col] = (_Float16)(v * v); }
      }
    }
  }
}

// ---------------------------------------------------------------------------
// Per-head RMSNorm + RoPE (+gain, +1/sqrt(HD) for Q) -> f16 head-major layout.
// 1 wave per (s, head); thread t owns elements t and t+32 (x1/x2 rope halves).
// ---------------------------------------------------------------------------
__global__ __launch_bounds__(32) void rope_qk(const float* __restrict__ in,
                                              _Float16* __restrict__ out,
                                              const float* __restrict__ gain,
                                              int S, int stride_in, float extra_scale) {
  int s = blockIdx.x, h = blockIdx.y, t = threadIdx.x;
  const float* p = in + (size_t)s * stride_in + h * 64;
  float x1 = p[t], x2 = p[t + 32];
  float ss = x1 * x1 + x2 * x2;
  for (int d = 1; d < 32; d <<= 1) ss += __shfl_xor(ss, d, 32);
  float r = rsqrtf(ss / 64.f + 1.1920929e-7f);
  x1 *= r; x2 *= r;
  float inv = __powf(10000.f, -(float)t / 32.f);
  float f = (float)s * inv;
  float sn, c;
  __sincosf(f, &sn, &c);
  float o1 = x1 * c + x2 * sn;
  float o2 = -x1 * sn + x2 * c;
  float g = extra_scale * (gain ? gain[h] : 1.f);
  _Float16* q = out + ((size_t)h * S + s) * 64;
  q[t]      = (_Float16)(o1 * g);
  q[t + 32] = (_Float16)(o2 * g);
}

// V: [S, KV*64] f32 -> vT16 [KV][64][S] f16 (so PV B-fragments are contiguous)
__global__ void vtrans(const float* __restrict__ vf, _Float16* __restrict__ vT,
                       int S, int KV) {
  int idx = blockIdx.x * blockDim.x + threadIdx.x;
  int total = KV * 64 * S;
  if (idx < total) {
    int s = idx % S; int rest = idx / S;
    int hd = rest % 64; int kv = rest / 64;
    vT[idx] = (_Float16)vf[(size_t)s * (KV * 64) + kv * 64 + hd];
  }
}

// ---------------------------------------------------------------------------
// Flash-style causal attention, 1 wave per (head, 16-query tile).
// QK^T + online softmax + PV, all on v_wmma_f32_16x16x32_f16.
// ---------------------------------------------------------------------------
__global__ __launch_bounds__(32) void attn_kernel(const _Float16* __restrict__ q16,
                                                  const _Float16* __restrict__ k16,
                                                  const _Float16* __restrict__ vT16,
                                                  _Float16* __restrict__ y16,
                                                  int S, int H, int KV) {
  __shared__ __align__(16) _Float16 lp[16 * 40];   // P tile, padded stride
  int lane = threadIdx.x;
  int ln = lane & 15, lh = lane >> 4;
  int qbase = blockIdx.x * 16;
  int h = blockIdx.y;
  int g = h / (H / KV);

  const _Float16* qrow = q16 + ((size_t)h * S + qbase + ln) * 64;
  v16h aq0 = mk16(qrow + 8 * lh, qrow + 16 + 8 * lh);               // hd 0..31
  v16h aq1 = mk16(qrow + 32 + 8 * lh, qrow + 48 + 8 * lh);          // hd 32..63

  v8f zero = {};
  v8f o[4];
  float rmax[8], rsum[8];
#pragma unroll
  for (int t = 0; t < 4; ++t) o[t] = zero;
#pragma unroll
  for (int i = 0; i < 8; ++i) { rmax[i] = -3.0e38f; rsum[i] = 0.f; }

  int nkt = (qbase + 16 + 31) / 32;
  for (int kt = 0; kt < nkt; ++kt) {
    int kbase = kt * 32;
    v8f sc[2];
#pragma unroll
    for (int sub = 0; sub < 2; ++sub) {
      int key = kbase + sub * 16 + ln;
      const _Float16* krow = k16 + ((size_t)g * S + key) * 64;
      v16h b0 = mk16(krow + 8 * lh, krow + 16 + 8 * lh);           // hd 0..31
      v16h b1 = mk16(krow + 32 + 8 * lh, krow + 48 + 8 * lh); // hd 32..63
      v8f t = zero;
      t = WMMA16(aq0, b0, t);
      t = WMMA16(aq1, b1, t);
      sc[sub] = t;
    }
    // causal mask + online softmax (row stats replicated across 16-lane halves)
#pragma unroll
    for (int i = 0; i < 8; ++i) {
      int q = qbase + i + 8 * lh;
      if (kbase + ln > q)      sc[0][i] = -3.0e38f;
      if (kbase + 16 + ln > q) sc[1][i] = -3.0e38f;
      float m = fmaxf(sc[0][i], sc[1][i]);
      for (int d = 1; d < 16; d <<= 1) m = fmaxf(m, __shfl_xor(m, d, 32));
      float nm = fmaxf(rmax[i], m);
      float alpha = __expf(rmax[i] - nm);
      float p0 = __expf(sc[0][i] - nm);
      float p1 = __expf(sc[1][i] - nm);
      float rs = p0 + p1;
      for (int d = 1; d < 16; d <<= 1) rs += __shfl_xor(rs, d, 32);
      rsum[i] = rsum[i] * alpha + rs;
      rmax[i] = nm;
#pragma unroll
      for (int t = 0; t < 4; ++t) o[t][i] *= alpha;
      int row = i + 8 * lh;
      lp[row * 40 + ln]      = (_Float16)p0;
      lp[row * 40 + 16 + ln] = (_Float16)p1;
    }
    asm volatile("s_wait_dscnt 0" ::: "memory");
    // P as A-fragment: lane M=ln, K chunks {8*lh..+7} and {16+8*lh..+7}
    v16h ap = mk16(&lp[ln * 40 + 8 * lh], &lp[ln * 40 + 16 + 8 * lh]);
#pragma unroll
    for (int t = 0; t < 4; ++t) {
      const _Float16* vrow = vT16 + ((size_t)g * 64 + t * 16 + ln) * S + kbase + 8 * lh;
      v16h bv = mk16(vrow, vrow + 16);
      o[t] = WMMA16(ap, bv, o[t]);
    }
  }
  // normalize + write y (f16, [S, H*64])
#pragma unroll
  for (int t = 0; t < 4; ++t) {
#pragma unroll
    for (int i = 0; i < 8; ++i) {
      int row = qbase + i + 8 * lh;
      float v = o[t][i] / rsum[i];
      y16[(size_t)row * (H * 64) + h * 64 + t * 16 + ln] = (_Float16)v;
    }
  }
}

// ---------------------------------------------------------------------------
// Host orchestration
// ---------------------------------------------------------------------------
extern "C" void kernel_launch(void* const* d_in, const int* in_sizes, int n_in,
                              void* d_out, int out_size, void* d_ws, size_t ws_size,
                              hipStream_t stream) {
  (void)in_sizes; (void)n_in; (void)out_size; (void)ws_size;
  const int S = 1024, D = 1024, H = 16, KV = 4, HD = 64, FF = 4096, V = 50257, NL = 8, KVD = 256;

  const int*   idx = (const int*)d_in[0];
  const float* emb = (const float*)d_in[1];
  const float* wq  = (const float*)d_in[2];
  const float* wk  = (const float*)d_in[3];
  const float* wv  = (const float*)d_in[4];
  const float* wo  = (const float*)d_in[5];
  const float* qg  = (const float*)d_in[6];
  const float* wfc = (const float*)d_in[7];
  const float* wpr = (const float*)d_in[8];
  const float* lmh = (const float*)d_in[9];
  float* out = (float*)d_out;

  char* base = (char*)d_ws;
  size_t off = 0;
  auto alloc = [&](size_t bytes) -> void* {
    off = (off + 255) & ~(size_t)255;
    void* p = base + off;
    off += bytes;
    return p;
  };
  float*    x    = (float*)alloc((size_t)S * D * 4);
  _Float16* h16  = (_Float16*)alloc((size_t)S * D * 2);
  float*    qf   = (float*)alloc((size_t)S * D * 4);
  float*    kf   = (float*)alloc((size_t)S * KVD * 4);
  float*    vf   = (float*)alloc((size_t)S * KVD * 4);
  _Float16* q16  = (_Float16*)alloc((size_t)S * D * 2);
  _Float16* k16  = (_Float16*)alloc((size_t)KV * S * HD * 2);
  _Float16* vT16 = (_Float16*)alloc((size_t)KV * HD * S * 2);
  _Float16* y16  = (_Float16*)alloc((size_t)S * D * 2);
  _Float16* f16b = (_Float16*)alloc((size_t)S * FF * 2);
  _Float16* wbuf = (_Float16*)alloc((size_t)V * D * 2);

  embed_kernel<<<dim3((S * D + 255) / 256), 256, 0, stream>>>(idx, emb, x, S, D);

  for (int l = 0; l < NL; ++l) {
    rmsnorm_f16<<<S, 256, 0, stream>>>(x, h16, D);

    quant_rows<<<D, 256, 0, stream>>>(wq + (size_t)l * D * D, wbuf, D);
    gemm_wmma<<<dim3(D / 64, S / 128), 256, 0, stream>>>(h16, wbuf, qf, nullptr, S, D, D, 0);
    quant_rows<<<KVD, 256, 0, stream>>>(wk + (size_t)l * KVD * D, wbuf, D);
    gemm_wmma<<<dim3((KVD + 63) / 64, S / 128), 256, 0, stream>>>(h16, wbuf, kf, nullptr, S, KVD, D, 0);
    quant_rows<<<KVD, 256, 0, stream>>>(wv + (size_t)l * KVD * D, wbuf, D);
    gemm_wmma<<<dim3((KVD + 63) / 64, S / 128), 256, 0, stream>>>(h16, wbuf, vf, nullptr, S, KVD, D, 0);

    rope_qk<<<dim3(S, H), 32, 0, stream>>>(qf, q16, qg + (size_t)l * H, S, D, 0.125f);   // gain * 1/sqrt(64)
    rope_qk<<<dim3(S, KV), 32, 0, stream>>>(kf, k16, nullptr, S, KVD, 1.0f);
    vtrans<<<(KV * HD * S + 255) / 256, 256, 0, stream>>>(vf, vT16, S, KV);

    attn_kernel<<<dim3(S / 16, H), 32, 0, stream>>>(q16, k16, vT16, y16, S, H, KV);

    quant_rows<<<D, 256, 0, stream>>>(wo + (size_t)l * D * D, wbuf, D);
    gemm_wmma<<<dim3(D / 64, S / 128), 256, 0, stream>>>(y16, wbuf, x, nullptr, S, D, D, 1);   // x += y@wo^T

    rmsnorm_f16<<<S, 256, 0, stream>>>(x, h16, D);
    quant_rows<<<FF, 256, 0, stream>>>(wfc + (size_t)l * FF * D, wbuf, D);
    gemm_wmma<<<dim3(FF / 64, S / 128), 256, 0, stream>>>(h16, wbuf, nullptr, f16b, S, FF, D, 2); // relu^2
    quant_rows<<<D, 256, 0, stream>>>(wpr + (size_t)l * D * FF, wbuf, FF);
    gemm_wmma<<<dim3(D / 64, S / 128), 256, 0, stream>>>(f16b, wbuf, x, nullptr, S, D, FF, 1);  // x += ff@proj^T
  }

  rmsnorm_f16<<<S, 256, 0, stream>>>(x, h16, D);
  quant_rows<<<V, 256, 0, stream>>>(lmh, wbuf, D);
  gemm_wmma<<<dim3((V + 63) / 64, S / 128), 256, 0, stream>>>(h16, wbuf, out, nullptr, S, V, D, 0);
}
